// GAT_54159537602882
// MI455X (gfx1250) — matrix operand order repaired and checked by reading, not compiled, not run on hardware
//
#include <hip/hip_runtime.h>
#include <math.h>

#define BB 2
#define NN 256
#define HH 2
#define FF 768     // Fin / Fout (nhid) for both layers
#define EO 384     // edge hidden (ehid)
#define E0 7       // raw edge features

typedef __attribute__((ext_vector_type(16))) __bf16 v16bf;
typedef __attribute__((ext_vector_type(8)))  __bf16 v8bf;
typedef __attribute__((ext_vector_type(8)))  float  v8f;

// ---------------- WMMA helpers (wave32, 16x16x32 bf16 -> f32) ----------------

__device__ __forceinline__ v8f wmma_bf16(v16bf a, v16bf b, v8f c) {
  return __builtin_amdgcn_wmma_f32_16x16x32_bf16(
      /*neg_a=*/false, a, /*neg_b=*/false, b,
      /*c_mod=*/(short)0, c, /*reuse_a=*/false, /*reuse_b=*/false);
}

// A fragment (16x32 tile, row-major f32 in global, converted to bf16).
// lanes 0-15: row=lane, K = k0+{0..7} and k0+{16..23}
// lanes16-31: row=lane-16, K = k0+{8..15} and k0+{24..31}
__device__ __forceinline__ v16bf a_frag_global(const float* __restrict__ A, int ld,
                                               int m0, int k0, int lane) {
  int row = m0 + (lane & 15);
  int kA  = k0 + ((lane >> 4) << 3);
  const float4* q = (const float4*)(A + (size_t)row * ld + kA);
  float4 x0 = q[0], x1 = q[1], x2 = q[4], x3 = q[5];
  v16bf a;
  a[0]=(__bf16)x0.x; a[1]=(__bf16)x0.y; a[2]=(__bf16)x0.z;  a[3]=(__bf16)x0.w;
  a[4]=(__bf16)x1.x; a[5]=(__bf16)x1.y; a[6]=(__bf16)x1.z;  a[7]=(__bf16)x1.w;
  a[8]=(__bf16)x2.x; a[9]=(__bf16)x2.y; a[10]=(__bf16)x2.z; a[11]=(__bf16)x2.w;
  a[12]=(__bf16)x3.x;a[13]=(__bf16)x3.y;a[14]=(__bf16)x3.z; a[15]=(__bf16)x3.w;
  return a;
}

// A fragment from an LDS bf16 stage (row stride in halves; 16B aligned rows).
__device__ __forceinline__ v16bf a_frag_lds(const __bf16* As, int stride,
                                            int k0, int lane) {
  int row = lane & 15;
  int kA  = k0 + ((lane >> 4) << 3);
  const __bf16* p = As + row * stride + kA;
  v8bf lo = *(const v8bf*)p;
  v8bf hi = *(const v8bf*)(p + 16);
  v16bf a;
#pragma unroll
  for (int j = 0; j < 8; ++j) { a[j] = lo[j]; a[8 + j] = hi[j]; }
  return a;
}

// B fragment from pre-packed buffer: tile (kt,nt) holds 512 bf16,
// lane's 16 values contiguous (one 32B load).
__device__ __forceinline__ v16bf b_frag(const __bf16* Bp, int tilesN,
                                        int kt, int nt, int lane) {
  const __bf16* p = Bp + (((size_t)(kt * tilesN + nt) * 32 + lane) << 4);
  return *(const v16bf*)p;
}

__device__ __forceinline__ float eluf(float x)  { return x > 0.f ? x : __expf(x) - 1.f; }

// ---------------- pack f32 [K x Nc] into bf16 B-operand layout ---------------

__global__ void pack_b(const float* __restrict__ src, void* dstv, int K, int Nc) {
  __bf16* dst = (__bf16*)dstv;
  int gid = blockIdx.x * blockDim.x + threadIdx.x;
  int total = (K / 32) * (Nc / 16) * 32;
  if (gid >= total) return;
  int t = gid >> 5, l = gid & 31;
  int tilesN = Nc / 16;
  int kt = t / tilesN, nt = t % tilesN;
  int c  = nt * 16 + (l & 15);
  int kb = kt * 32 + ((l >> 4) << 3);     // lanes 0-15: K k0+{0..7,16..23}, 16-31: +8
  __bf16* d = dst + ((size_t)t << 9) + (l << 4);
#pragma unroll
  for (int j = 0; j < 16; ++j) d[j] = (__bf16)src[(size_t)(kb + j + ((j >> 3) << 3)) * Nc + c];
}

// ---------------- v[h] = We[h] @ a_e[h]  (rows = H*Ein, reduce over Eout) ----

__global__ void vedge(const float* __restrict__ We, const float* __restrict__ ae,
                      float* __restrict__ v, int Ein, int Eout) {
  int wid  = (blockIdx.x * blockDim.x + threadIdx.x) >> 5;
  int lane = threadIdx.x & 31;
  if (wid >= HH * Ein) return;
  int h = wid / Ein;
  const float* wr = We + (size_t)wid * Eout;
  const float* a  = ae + (size_t)h * Eout;
  float s = 0.f;
  for (int j = lane; j < Eout; j += 32) s += wr[j] * a[j];
  for (int off = 16; off; off >>= 1) s += __shfl_down(s, off, 32);
  if (lane == 0) v[wid] = s;
}

// ---------------- Wh[b,h] = X[b] @ W[h]   (M=256, K=768, N=768) --------------

__global__ __launch_bounds__(256) void gemm_xw(const float* __restrict__ X,
                                               const void* Wpv, float* __restrict__ Wh) {
  const __bf16* Wp = (const __bf16*)Wpv;
  int wid  = (blockIdx.x * blockDim.x + threadIdx.x) >> 5;
  int lane = threadIdx.x & 31;
  const int TN = FF / 16;                     // 48
  int nt = wid % TN;
  int mt = (wid / TN) % (NN / 16);
  int h  = (wid / (TN * (NN / 16))) % HH;
  int b  =  wid / (TN * (NN / 16) * HH);
  const float*  A  = X  + (size_t)b * NN * FF;
  const __bf16* Bp = Wp + (size_t)h * FF * FF;
  v8f acc = {};
  for (int kt = 0; kt < FF / 32; ++kt) {
    v16bf a  = a_frag_global(A, FF, mt * 16, kt * 32, lane);
    v16bf bb = b_frag(Bp, TN, kt, nt, lane);
    acc = wmma_bf16(a, bb, acc);
  }
  float* out = Wh + (((size_t)(b * HH + h) * NN + mt * 16) * FF) + nt * 16 + (lane & 15);
  int ro = (lane >> 4) * 8;
#pragma unroll
  for (int r = 0; r < 8; ++r) out[(size_t)(ro + r) * FF] = acc[r];
}

// ---------------- s_src / s_dst: dot(Wh[b,h,n,:], a)  ------------------------

__global__ void sdots(const float* __restrict__ Wh, const float* __restrict__ as,
                      const float* __restrict__ ad, float* __restrict__ ssrc,
                      float* __restrict__ sdst) {
  int wid  = (blockIdx.x * blockDim.x + threadIdx.x) >> 5;   // (b*H+h)*N + n
  int lane = threadIdx.x & 31;
  int h = (wid / NN) % HH;
  const float* row = Wh + (size_t)wid * FF;
  const float* va = as + (size_t)h * FF;
  const float* vd = ad + (size_t)h * FF;
  float s0 = 0.f, s1 = 0.f;
  for (int o = lane; o < FF; o += 32) { float w = row[o]; s0 += w * va[o]; s1 += w * vd[o]; }
  for (int off = 16; off; off >>= 1) { s0 += __shfl_down(s0, off, 32); s1 += __shfl_down(s1, off, 32); }
  if (lane == 0) { ssrc[wid] = s0; sdst[wid] = s1; }
}

// ------------- scores = leaky(ssrc+sdst+sedge), mask, softmax ----------------
// e7 != null  -> layer1: s_edge computed inline as 7-dot with v1
// e7 == null  -> layer2: s_edge read from buffer (produced by eprime1)

__global__ __launch_bounds__(NN) void softmax_att(
    const int* __restrict__ adj, const float* __restrict__ ssrc,
    const float* __restrict__ sdst, const float* __restrict__ e7,
    const float* __restrict__ v1, const float* __restrict__ sedge,
    float* __restrict__ att) {
  int bh = blockIdx.x / NN;        // b*H + h
  int n  = blockIdx.x % NN;
  int b  = bh / HH, h = bh % HH;
  int m  = threadIdx.x;
  float se;
  if (e7) {
    const float* ep = e7 + ((size_t)(b * NN + n) * NN + m) * E0;
    float acc = 0.f;
#pragma unroll
    for (int i = 0; i < E0; ++i) acc += ep[i] * v1[h * E0 + i];
    se = acc;
  } else {
    se = sedge[((size_t)bh * NN + n) * NN + m];
  }
  float s = ssrc[bh * NN + n] + sdst[bh * NN + m] + se;
  s = s > 0.f ? s : 0.2f * s;                      // leaky_relu(0.2)
  bool on = adj[((size_t)b * NN + n) * NN + m] > 0;
  float sc = on ? s : -9e15f;
  __shared__ float red[NN];
  red[m] = sc; __syncthreads();
  for (int st = NN / 2; st > 0; st >>= 1) { if (m < st) red[m] = fmaxf(red[m], red[m + st]); __syncthreads(); }
  float mx = red[0]; __syncthreads();
  float ex = on ? __expf(sc - mx) : 0.f;
  red[m] = ex; __syncthreads();
  for (int st = NN / 2; st > 0; st >>= 1) { if (m < st) red[m] += red[m + st]; __syncthreads(); }
  float inv = 1.f / red[0];
  att[((size_t)bh * NN + n) * NN + m] = ex * inv;
}

// ---- x' = 0.5*(elu(att0@Wh0)+elu(att1@Wh1)) + res   (M=256,K=256,N=768) -----

__global__ __launch_bounds__(256) void hprime(const float* __restrict__ att,
                                              const void* Whpv,
                                              const float* __restrict__ res,
                                              float* __restrict__ xout) {
  const __bf16* Whp = (const __bf16*)Whpv;
  int wid  = (blockIdx.x * blockDim.x + threadIdx.x) >> 5;
  int lane = threadIdx.x & 31;
  const int TN = FF / 16;                       // 48
  int nt = wid % TN;
  int mt = (wid / TN) % (NN / 16);
  int b  =  wid / (TN * (NN / 16));
  const float*  A0 = att + (size_t)(b * HH + 0) * NN * NN;
  const float*  A1 = att + (size_t)(b * HH + 1) * NN * NN;
  const __bf16* B0 = Whp + (size_t)(b * HH + 0) * NN * FF;
  const __bf16* B1 = Whp + (size_t)(b * HH + 1) * NN * FF;
  v8f acc0 = {}, acc1 = {};
  for (int kt = 0; kt < NN / 32; ++kt) {
    v16bf a0 = a_frag_global(A0, NN, mt * 16, kt * 32, lane);
    acc0 = wmma_bf16(a0, b_frag(B0, TN, kt, nt, lane), acc0);
    v16bf a1 = a_frag_global(A1, NN, mt * 16, kt * 32, lane);
    acc1 = wmma_bf16(a1, b_frag(B1, TN, kt, nt, lane), acc1);
  }
  int col = nt * 16 + (lane & 15);
  int ro  = (lane >> 4) * 8;
#pragma unroll
  for (int r = 0; r < 8; ++r) {
    int nrow = mt * 16 + ro + r;
    size_t idx = ((size_t)b * NN + nrow) * FF + col;
    xout[idx] = 0.5f * (eluf(acc0[r]) + eluf(acc1[r])) + res[idx];
  }
}

// ------- e1 = 0.5*sum_h att1_h * (e @ We1[h])  (K=7 -> VALU, store-bound) ----
// Fused: also emits layer-2 edge scores  sed[b,h,n,m] = e1[b,n,m,:] . v2[h]
// while the freshly computed e1 row is still in registers (saves a full
// 201 MB re-read of e1 in layer 2).

__global__ void eprime1(const float* __restrict__ e, const float* __restrict__ We1,
                        const float* __restrict__ att, const float* __restrict__ v2,
                        float* __restrict__ e1out, float* __restrict__ sedge) {
  __shared__ float LWe[HH * E0 * EO];           // 5376 f32
  __shared__ float v2L[2 * EO];                 // 768 f32
  for (int i = threadIdx.x; i < HH * E0 * EO; i += blockDim.x) LWe[i] = We1[i];
  for (int i = threadIdx.x; i < 2 * EO; i += blockDim.x) v2L[i] = v2[i];
  __syncthreads();
  int wid = threadIdx.x >> 5, lane = threadIdx.x & 31;
  for (int t = 0; t < 16; ++t) {
    int edge = (blockIdx.x * 8 + wid) * 16 + t;  // < B*N*N
    const float* ep = e + (size_t)edge * E0;
    if (t < 15) __builtin_prefetch(ep + E0, 0, 0);   // global_prefetch_b8 next row
    float c[E0];
#pragma unroll
    for (int i = 0; i < E0; ++i) c[i] = ep[i];
    int b = edge >> 16, nm = edge & 65535;
    float a0 = att[(size_t)(b * HH + 0) * NN * NN + nm];
    float a1 = att[(size_t)(b * HH + 1) * NN * NN + nm];
    float* orow = e1out + (size_t)edge * EO;
    float se0 = 0.f, se1 = 0.f;
    for (int j = lane; j < EO; j += 32) {
      float s0 = 0.f, s1 = 0.f;
#pragma unroll
      for (int i = 0; i < E0; ++i) {
        s0 += c[i] * LWe[i * EO + j];
        s1 += c[i] * LWe[(E0 + i) * EO + j];
      }
      float val = 0.5f * (a0 * s0 + a1 * s1);
      orow[j] = val;
      se0 += val * v2L[j];
      se1 += val * v2L[EO + j];
    }
    for (int off = 16; off; off >>= 1) {
      se0 += __shfl_down(se0, off, 32);
      se1 += __shfl_down(se1, off, 32);
    }
    if (lane == 0) {
      sedge[(size_t)(b * HH + 0) * NN * NN + nm] = se0;
      sedge[(size_t)(b * HH + 1) * NN * NN + nm] = se1;
    }
  }
}

// -- e2 = 0.5*sum_h att2_h * (e1 @ We2[h]), in place over e1 (WMMA, LDS A) ----

__global__ __launch_bounds__(128) void eprime2(float* __restrict__ ebuf,
                                               const void* We2pv,
                                               const float* __restrict__ att) {
  const __bf16* We2p = (const __bf16*)We2pv;
  const int STR = EO + 8;                      // 392 halves: kill bank conflicts
  __shared__ __align__(16) __bf16 As[16 * STR];
  __shared__ float attL[2][16];
  int bid = blockIdx.x;                        // (b*256 + n)*16 + mtile
  int mt = bid & 15;
  int n  = (bid >> 4) & 255;
  int b  =  bid >> 12;
  int m0 = mt * 16;
  float* rows = ebuf + ((size_t)(b * NN + n) * NN + m0) * EO;
  // stage 16 full rows of e1 as bf16 (all global reads before any write)
  for (int i = threadIdx.x; i < 16 * EO; i += blockDim.x) {
    int r = i / EO, c = i % EO;
    As[r * STR + c] = (__bf16)rows[(size_t)r * EO + c];
  }
  if (threadIdx.x < 32) {
    int h = threadIdx.x >> 4, mm = threadIdx.x & 15;
    attL[h][mm] = att[(size_t)(b * HH + h) * NN * NN + (size_t)n * NN + m0 + mm];
  }
  __syncthreads();
  int wid = threadIdx.x >> 5, lane = threadIdx.x & 31;
  const int TN = EO / 16;                      // 24 col tiles
  for (int ct = wid; ct < TN; ct += 4) {
    v8f acc0 = {}, acc1 = {};
    for (int kt = 0; kt < EO / 32; ++kt) {
      v16bf a = a_frag_lds(As, STR, kt * 32, lane);
      acc0 = wmma_bf16(a, b_frag(We2p,           TN, kt, ct, lane), acc0);
      acc1 = wmma_bf16(a, b_frag(We2p + EO * EO, TN, kt, ct, lane), acc1);
    }
    int col = ct * 16 + (lane & 15);
    int ro  = (lane >> 4) * 8;
#pragma unroll
    for (int r = 0; r < 8; ++r) {
      int rr = ro + r;
      rows[(size_t)rr * EO + col] = 0.5f * (attL[0][rr] * acc0[r] + attL[1][rr] * acc1[r]);
    }
  }
}

// ---------------------------------- host -------------------------------------

extern "C" void kernel_launch(void* const* d_in, const int* in_sizes, int n_in,
                              void* d_out, int out_size, void* d_ws, size_t ws_size,
                              hipStream_t stream) {
  (void)in_sizes; (void)n_in; (void)out_size; (void)ws_size;
  const float* x   = (const float*)d_in[0];
  const int*   adj = (const int*)  d_in[1];
  const float* e   = (const float*)d_in[2];
  const float* W1  = (const float*)d_in[3];
  const float* We1 = (const float*)d_in[4];
  const float* a1s = (const float*)d_in[5];
  const float* a1d = (const float*)d_in[6];
  const float* a1e = (const float*)d_in[7];
  const float* W2  = (const float*)d_in[8];
  const float* We2 = (const float*)d_in[9];
  const float* a2s = (const float*)d_in[10];
  const float* a2d = (const float*)d_in[11];
  const float* a2e = (const float*)d_in[12];

  char* ws = (char*)d_ws;
  size_t off = 0;
  auto take = [&](size_t bytes) -> char* {
    char* p = ws + off; off = (off + bytes + 255) & ~(size_t)255; return p;
  };
  float*          Wh   = (float*)         take((size_t)BB * HH * NN * FF * 4);
  unsigned short* Whp  = (unsigned short*)take((size_t)BB * HH * NN * FF * 2);
  unsigned short* W1p  = (unsigned short*)take((size_t)HH * FF * FF * 2);
  unsigned short* W2p  = (unsigned short*)take((size_t)HH * FF * FF * 2);
  unsigned short* We2p = (unsigned short*)take((size_t)HH * EO * EO * 2);
  float*          att  = (float*)take((size_t)BB * HH * NN * NN * 4);
  float*          sed  = (float*)take((size_t)BB * HH * NN * NN * 4);
  float*          ssrc = (float*)take((size_t)BB * HH * NN * 4);
  float*          sdst = (float*)take((size_t)BB * HH * NN * 4);
  float*          v1   = (float*)take((size_t)HH * E0 * 4);
  float*          v2   = (float*)take((size_t)HH * EO * 4);
  float*          x1   = (float*)take((size_t)BB * NN * FF * 4);

  float* xout = (float*)d_out;                      // [B,N,768]
  float* eout = xout + (size_t)BB * NN * FF;        // [B,N,N,384]; holds e1, then e2 in place

  // ---- static weight packs + edge-score vectors ----
  for (int h = 0; h < HH; ++h) {
    pack_b<<<144, 256, 0, stream>>>(W1 + (size_t)h * FF * FF, W1p + (size_t)h * FF * FF, FF, FF);
    pack_b<<<144, 256, 0, stream>>>(W2 + (size_t)h * FF * FF, W2p + (size_t)h * FF * FF, FF, FF);
    pack_b<<<36,  256, 0, stream>>>(We2 + (size_t)h * EO * EO, We2p + (size_t)h * EO * EO, EO, EO);
  }
  vedge<<<2,  256, 0, stream>>>(We1, a1e, v1, E0, EO);
  vedge<<<96, 256, 0, stream>>>(We2, a2e, v2, EO, EO);

  // ---- layer 1 ----
  gemm_xw<<<384, 256, 0, stream>>>(x, W1p, Wh);
  sdots<<<128, 256, 0, stream>>>(Wh, a1s, a1d, ssrc, sdst);
  softmax_att<<<BB * HH * NN, NN, 0, stream>>>(adj, ssrc, sdst, e, v1, nullptr, att);
  for (int i = 0; i < BB * HH; ++i)
    pack_b<<<48, 256, 0, stream>>>(Wh + (size_t)i * NN * FF, Whp + (size_t)i * NN * FF, NN, FF);
  hprime<<<192, 256, 0, stream>>>(att, Whp, x, x1);
  // e1 -> d_out e-region; also emits layer-2 s_edge on the fly
  eprime1<<<1024, 256, 0, stream>>>(e, We1, att, v2, eout, sed);

  // ---- layer 2 ----
  gemm_xw<<<384, 256, 0, stream>>>(x1, W2p, Wh);
  sdots<<<128, 256, 0, stream>>>(Wh, a2s, a2d, ssrc, sdst);
  softmax_att<<<BB * HH * NN, NN, 0, stream>>>(adj, ssrc, sdst, nullptr, nullptr, sed, att);
  for (int i = 0; i < BB * HH; ++i)
    pack_b<<<48, 256, 0, stream>>>(Wh + (size_t)i * NN * FF, Whp + (size_t)i * NN * FF, NN, FF);
  hprime<<<192, 256, 0, stream>>>(att, Whp, x, xout);            // x2
  eprime2<<<8192, 128, 0, stream>>>(eout, We2p, att);            // e2 in place
}
